// GaussianLayer_80642305950189
// MI455X (gfx1250) — compile-verified, matrix operand order checked by reading, not run on hardware
//
#include <hip/hip_runtime.h>
#include <cstdint>
#include <cstddef>

// ---------------------------------------------------------------------------
// GaussianLayer on MI455X (gfx1250, wave32, WMMA + TDM).
//
// acc[m,n] = -(z*z)@d^T + z@(2*d*mean)^T     (2 chained WMMAs per K-step)
// out[m,n] = tanh(scale[n]) * exp(acc[m,n] - mcm[n])
//
// M=16384, K=1024, N=2048.  bf16 WMMA 16x16x32, f32 accumulate.
//
// v3.1: B tiles (shared by all 8 waves of a block) staged in LDS by the
// Tensor Data Mover, double-buffered so the DMA for K-step i+1 overlaps the
// WMMAs of step i.  This toolchain exposes the 6-arg tensor_load_to_lds
// builtin (clang-23 / therock-10.0): (u32x4, i32x8, i32x4, i32x4, i32x8, i32).
// ---------------------------------------------------------------------------

typedef __attribute__((ext_vector_type(16))) __bf16 v16bf;
typedef __attribute__((ext_vector_type(8)))  __bf16 v8bf;
typedef __attribute__((ext_vector_type(8)))  float  v8f;
typedef __attribute__((ext_vector_type(4)))  float  v4f;
typedef __attribute__((ext_vector_type(4)))  unsigned int u32x4;
typedef __attribute__((ext_vector_type(8)))  int   i32x8;
typedef __attribute__((ext_vector_type(4)))  int   i32x4;

static constexpr int M_ROWS = 16384;
static constexpr int IN_F   = 1024;   // K
static constexpr int OUT_F  = 2048;   // N

#if __has_builtin(__builtin_amdgcn_tensor_load_to_lds) && \
    __has_builtin(__builtin_amdgcn_s_wait_tensorcnt)
#define USE_TDM 1
#else
#define USE_TDM 0
#endif

// float -> bf16, round-to-nearest-even.
__device__ __forceinline__ __bf16 f2bf(float f) {
    uint32_t u = __float_as_uint(f);
    uint32_t r = (u + 0x7FFFu + ((u >> 16) & 1u)) >> 16;
    unsigned short s = (unsigned short)r;
    return __builtin_bit_cast(__bf16, s);
}

// ---------------------------------------------------------------------------
// Prep B: one block per output feature k (2048 blocks x 256 threads).
// ---------------------------------------------------------------------------
__global__ void __launch_bounds__(256)
gauss_prep(const float* __restrict__ diag,
           const float* __restrict__ mean,
           const float* __restrict__ scale,
           __bf16* __restrict__ Bneg,
           __bf16* __restrict__ Bmean,
           float*  __restrict__ mcm,
           float*  __restrict__ tanhs) {
    __shared__ float red[256];
    const int k = blockIdx.x;
    const int t = threadIdx.x;

    float local = 0.0f;
#pragma unroll
    for (int j = 0; j < IN_F / 256; ++j) {
        const int i  = t + j * 256;
        const size_t idx = (size_t)k * IN_F + i;
        const float dp = diag[idx];
        const float d  = 0.1f + 0.9f / (1.0f + __expf(-dp));   // sigmoid clamp
        const float mu = mean[idx];
        Bneg[idx]  = f2bf(-d);
        Bmean[idx] = f2bf(2.0f * d * mu);
        local += d * mu * mu;
    }
    red[t] = local;
    __syncthreads();
#pragma unroll
    for (int s = 128; s > 0; s >>= 1) {
        if (t < s) red[t] += red[t + s];
        __syncthreads();
    }
    if (t == 0) {
        mcm[k]   = red[0];
        tanhs[k] = tanhf(scale[k]);
    }
}

// ---------------------------------------------------------------------------
// Prep A: z (f32, row-major) -> Azz = bf16(z*z), Az = bf16(z).
// ---------------------------------------------------------------------------
__global__ void __launch_bounds__(256)
gauss_prep_a(const float* __restrict__ z,
             __bf16* __restrict__ Azz,
             __bf16* __restrict__ Az) {
    const size_t base = ((size_t)blockIdx.x * 256 + threadIdx.x) * 8;
    const v4f f0 = *(const v4f*)(z + base);
    const v4f f1 = *(const v4f*)(z + base + 4);
    v8bf vz, vzz;
#pragma unroll
    for (int e = 0; e < 4; ++e) {
        vz[e]      = f2bf(f0[e]);
        vz[4 + e]  = f2bf(f1[e]);
        vzz[e]     = f2bf(f0[e] * f0[e]);
        vzz[4 + e] = f2bf(f1[e] * f1[e]);
    }
    *(v8bf*)(Az  + base) = vz;
    *(v8bf*)(Azz + base) = vzz;
}

#if USE_TDM
// ---------------------------------------------------------------------------
// TDM: load one 64(N-rows) x 32(K) bf16 tile from a row-major (N x K=1024)
// matrix into LDS.  pad_enable adds 2 DWORDs after every 16 DWORDs (64B row)
// => LDS row stride 72B (36 bf16) to spread LDS banks.
//
// D# per CDNA5 ISA:
//  g0: [1:0]=count=1 | lds_addr | global_addr[56:0] | type=2
//  g1: data_size=1(2B), pad_enable, pad_interval=3(16 DW), pad_amount=1(2 DW),
//      tensor_dim0=1024, tensor_dim1=2048, tile_dim0=32, tile_dim1=64,
//      tensor_dim0_stride=1024
// ---------------------------------------------------------------------------
static constexpr int ROWSTRIDE = 36;   // bf16 elements per LDS row (32 + 4 pad)

__device__ __forceinline__ void tdm_load_tile(uint32_t lds_off, const __bf16* gptr) {
    const uint64_t ga = (uint64_t)(uintptr_t)gptr;
    u32x4 g0;
    g0[0] = 1u;                                              // count=1, user D#
    g0[1] = lds_off;                                         // lds_addr (bytes)
    g0[2] = (uint32_t)ga;                                    // global_addr[31:0]
    g0[3] = (uint32_t)((ga >> 32) & 0x1FFFFFFu) | (2u << 30); // addr[56:32]|type=2

    i32x8 g1;
    g1[0] = (int)((1u << 16) |            // data_size = 1 (2 bytes)
                  (1u << 20) |            // pad_enable
                  (3u << 22) |            // pad_interval: 16 DWORDs (64B)
                  (1u << 25));            // pad_amount: 2 DWORDs (8B)
    g1[1] = (int)(((uint32_t)IN_F & 0xFFFFu) << 16);          // tensor_dim0 lo16
    g1[2] = (int)((((uint32_t)IN_F >> 16) & 0xFFFFu) |
                  (((uint32_t)OUT_F & 0xFFFFu) << 16));       // dim0 hi | dim1 lo
    g1[3] = (int)((((uint32_t)OUT_F >> 16) & 0xFFFFu) |
                  (32u << 16));                               // dim1 hi | tile_dim0=32
    g1[4] = 64;                                               // tile_dim1=64, tile_dim2=0
    g1[5] = IN_F;                                             // tensor_dim0_stride lo32
    g1[6] = 0;                                                // stride hi | dim1_stride lo
    g1[7] = 0;

    i32x4 z4;
    z4[0] = 0; z4[1] = 0; z4[2] = 0; z4[3] = 0;
    i32x8 z8;
    z8[0] = 0; z8[1] = 0; z8[2] = 0; z8[3] = 0;
    z8[4] = 0; z8[5] = 0; z8[6] = 0; z8[7] = 0;
    // 6-arg form (clang-23 / therock-10.0 headers).
    __builtin_amdgcn_tensor_load_to_lds(g0, g1, z4, z4, z8, 0);
}

// ---------------------------------------------------------------------------
// TDM-staged GEMM.  Block = 256 threads (8 waves); wave w owns rows
// [(blockIdx.x*8+w)*16, +16), cols [blockIdx.y*64, +64).  B K-slabs are
// DMA'd to LDS (double-buffered) by wave 0, consumed via ds_load_b128.
// ---------------------------------------------------------------------------
__global__ void __launch_bounds__(256)
gauss_gemm_tdm(const __bf16* __restrict__ Azz,
               const __bf16* __restrict__ Az,
               const __bf16* __restrict__ Bneg,
               const __bf16* __restrict__ Bmean,
               const float*  __restrict__ mcm,
               const float*  __restrict__ tanhs,
               float* __restrict__ out) {
    __shared__ __bf16 lds[2][2][64 * ROWSTRIDE];   // [buf][matrix][row*ROWSTRIDE+k]

    const int lane = threadIdx.x & 31;
    const int wave = threadIdx.x >> 5;
    const int h    = lane >> 4;
    const int l16  = lane & 15;

    const int mTile = blockIdx.x * 8 + wave;
    const int nBase = blockIdx.y * 64;
    const size_t aOff = (size_t)(mTile * 16 + l16) * IN_F;

    v8f acc[4];
#pragma unroll
    for (int t = 0; t < 4; ++t) acc[t] = (v8f){};

    // Prologue: request K-step 0 into buffer 0.
    if (wave == 0) {
        tdm_load_tile((uint32_t)(uintptr_t)&lds[0][0][0], Bneg  + (size_t)nBase * IN_F);
        tdm_load_tile((uint32_t)(uintptr_t)&lds[0][1][0], Bmean + (size_t)nBase * IN_F);
    }

    constexpr int NSTEP = IN_F / 32;
    for (int i = 0; i < NSTEP; ++i) {
        const int k0 = i * 32;
        const int p  = i & 1;

        // Buffer p was requested last iteration (or prologue): drain TENSORcnt
        // on the issuing wave, then publish to the whole workgroup.
        if (wave == 0) __builtin_amdgcn_s_wait_tensorcnt((short)0);
        __syncthreads();

        // Prefetch next K-slab into the other buffer.  Issued after the
        // barrier, so no wave can still be reading buffer 1-p.
        if (wave == 0 && i + 1 < NSTEP) {
            const size_t gOff = (size_t)nBase * IN_F + (k0 + 32);
            tdm_load_tile((uint32_t)(uintptr_t)&lds[1 - p][0][0], Bneg  + gOff);
            tdm_load_tile((uint32_t)(uintptr_t)&lds[1 - p][1][0], Bmean + gOff);
        }

        // A operands (per-wave rows) straight from global bf16 workspace.
        const int o0 = k0 + 8 * h;
        const int o1 = o0 + 16;
        union U { v16bf v; v8bf q[2]; };
        U azz, az;
        azz.q[0] = *(const v8bf*)(Azz + aOff + o0);
        azz.q[1] = *(const v8bf*)(Azz + aOff + o1);
        az.q[0]  = *(const v8bf*)(Az  + aOff + o0);
        az.q[1]  = *(const v8bf*)(Az  + aOff + o1);

#pragma unroll
        for (int t = 0; t < 4; ++t) {
            const int row = t * 16 + l16;          // tile-local N row
            const __bf16* bnp = &lds[p][0][row * ROWSTRIDE];
            const __bf16* bmp = &lds[p][1][row * ROWSTRIDE];
            U bn, bm;
            bn.q[0] = *(const v8bf*)(bnp + 8 * h);
            bn.q[1] = *(const v8bf*)(bnp + 8 * h + 16);
            bm.q[0] = *(const v8bf*)(bmp + 8 * h);
            bm.q[1] = *(const v8bf*)(bmp + 8 * h + 16);

            acc[t] = __builtin_amdgcn_wmma_f32_16x16x32_bf16(
                false, azz.v, false, bn.v, (short)0, acc[t], false, false);
            acc[t] = __builtin_amdgcn_wmma_f32_16x16x32_bf16(
                false, az.v,  false, bm.v, (short)0, acc[t], false, false);
        }
    }

#pragma unroll
    for (int t = 0; t < 4; ++t) {
        const int n   = nBase + t * 16 + l16;
        const float s = tanhs[n];
        const float c = mcm[n];
#pragma unroll
        for (int r = 0; r < 8; ++r) {
            const int mrow = mTile * 16 + 8 * h + r;
            out[(size_t)mrow * OUT_F + n] = s * __expf(acc[t][r] - c);
        }
    }
}
#endif  // USE_TDM

// ---------------------------------------------------------------------------
// Pure-global WMMA GEMM (no TDM): used if the TDM builtin is unavailable.
// ---------------------------------------------------------------------------
__global__ void __launch_bounds__(256)
gauss_gemm_pre(const __bf16* __restrict__ Azz,
               const __bf16* __restrict__ Az,
               const __bf16* __restrict__ Bneg,
               const __bf16* __restrict__ Bmean,
               const float*  __restrict__ mcm,
               const float*  __restrict__ tanhs,
               float* __restrict__ out) {
    const int lane = threadIdx.x & 31;
    const int wave = threadIdx.x >> 5;
    const int h    = lane >> 4;
    const int l16  = lane & 15;

    const int mTile = blockIdx.x * 8 + wave;
    const int nBase = blockIdx.y * 64;
    const size_t aOff = (size_t)(mTile * 16 + l16) * IN_F;

    v8f acc[4];
#pragma unroll
    for (int t = 0; t < 4; ++t) acc[t] = (v8f){};

    for (int k0 = 0; k0 < IN_F; k0 += 32) {
        const int o0 = k0 + 8 * h;
        const int o1 = o0 + 16;

        union U { v16bf v; v8bf q[2]; };
        U azz, az;
        azz.q[0] = *(const v8bf*)(Azz + aOff + o0);
        azz.q[1] = *(const v8bf*)(Azz + aOff + o1);
        az.q[0]  = *(const v8bf*)(Az  + aOff + o0);
        az.q[1]  = *(const v8bf*)(Az  + aOff + o1);

#pragma unroll
        for (int t = 0; t < 4; ++t) {
            const size_t nrow = (size_t)(nBase + t * 16 + l16) * IN_F;
            U bn, bm;
            bn.q[0] = *(const v8bf*)(Bneg  + nrow + o0);
            bn.q[1] = *(const v8bf*)(Bneg  + nrow + o1);
            bm.q[0] = *(const v8bf*)(Bmean + nrow + o0);
            bm.q[1] = *(const v8bf*)(Bmean + nrow + o1);

            acc[t] = __builtin_amdgcn_wmma_f32_16x16x32_bf16(
                false, azz.v, false, bn.v, (short)0, acc[t], false, false);
            acc[t] = __builtin_amdgcn_wmma_f32_16x16x32_bf16(
                false, az.v,  false, bm.v, (short)0, acc[t], false, false);
        }
    }

#pragma unroll
    for (int t = 0; t < 4; ++t) {
        const int n   = nBase + t * 16 + l16;
        const float s = tanhs[n];
        const float c = mcm[n];
#pragma unroll
        for (int r = 0; r < 8; ++r) {
            const int mrow = mTile * 16 + 8 * h + r;
            out[(size_t)mrow * OUT_F + n] = s * __expf(acc[t][r] - c);
        }
    }
}

// ---------------------------------------------------------------------------
// Fallback GEMM: in-loop f32->bf16 A conversion (no A workspace needed).
// ---------------------------------------------------------------------------
__global__ void __launch_bounds__(256)
gauss_gemm_conv(const float*  __restrict__ z,
                const __bf16* __restrict__ Bneg,
                const __bf16* __restrict__ Bmean,
                const float*  __restrict__ mcm,
                const float*  __restrict__ tanhs,
                float* __restrict__ out) {
    const int lane = threadIdx.x & 31;
    const int wave = threadIdx.x >> 5;
    const int h    = lane >> 4;
    const int l16  = lane & 15;

    const int mTile = blockIdx.x * 8 + wave;
    const int nBase = blockIdx.y * 64;
    const float* aRow = z + (size_t)(mTile * 16 + l16) * IN_F;

    v8f acc[4];
#pragma unroll
    for (int t = 0; t < 4; ++t) acc[t] = (v8f){};

    for (int k0 = 0; k0 < IN_F; k0 += 32) {
        const int o0 = k0 + 8 * h;
        const int o1 = o0 + 16;

        const v4f a0 = *(const v4f*)(aRow + o0);
        const v4f a1 = *(const v4f*)(aRow + o0 + 4);
        const v4f a2 = *(const v4f*)(aRow + o1);
        const v4f a3 = *(const v4f*)(aRow + o1 + 4);
        float af[16];
#pragma unroll
        for (int e = 0; e < 4; ++e) {
            af[e] = a0[e]; af[4 + e] = a1[e]; af[8 + e] = a2[e]; af[12 + e] = a3[e];
        }
        v16bf azz, az;
#pragma unroll
        for (int e = 0; e < 16; ++e) {
            az[e]  = f2bf(af[e]);
            azz[e] = f2bf(af[e] * af[e]);
        }

#pragma unroll
        for (int t = 0; t < 4; ++t) {
            const size_t nrow = (size_t)(nBase + t * 16 + l16) * IN_F;
            union U { v16bf v; v8bf q[2]; } bn, bm;
            bn.q[0] = *(const v8bf*)(Bneg  + nrow + o0);
            bn.q[1] = *(const v8bf*)(Bneg  + nrow + o1);
            bm.q[0] = *(const v8bf*)(Bmean + nrow + o0);
            bm.q[1] = *(const v8bf*)(Bmean + nrow + o1);

            acc[t] = __builtin_amdgcn_wmma_f32_16x16x32_bf16(
                false, azz, false, bn.v, (short)0, acc[t], false, false);
            acc[t] = __builtin_amdgcn_wmma_f32_16x16x32_bf16(
                false, az,  false, bm.v, (short)0, acc[t], false, false);
        }
    }

#pragma unroll
    for (int t = 0; t < 4; ++t) {
        const int n   = nBase + t * 16 + l16;
        const float s = tanhs[n];
        const float c = mcm[n];
#pragma unroll
        for (int r = 0; r < 8; ++r) {
            const int mrow = mTile * 16 + 8 * h + r;
            out[(size_t)mrow * OUT_F + n] = s * __expf(acc[t][r] - c);
        }
    }
}

// ---------------------------------------------------------------------------
// Launch.  Workspace (full path, ~72.4 MB):
//   [0,4Mi) Bneg | [4Mi,8Mi) Bmean | mcm/tanhs (16K) | Azz 32Mi | Az 32Mi
// Kernel selection depends only on ws_size (fixed per run) => deterministic.
// ---------------------------------------------------------------------------
extern "C" void kernel_launch(void* const* d_in, const int* in_sizes, int n_in,
                              void* d_out, int out_size, void* d_ws, size_t ws_size,
                              hipStream_t stream) {
    (void)in_sizes; (void)n_in; (void)out_size;

    const float* z     = (const float*)d_in[0];  // (16384, 1024)
    const float* diag  = (const float*)d_in[1];  // (2048, 1024)
    const float* mean  = (const float*)d_in[2];  // (2048, 1024, 1)
    const float* scale = (const float*)d_in[3];  // (2048,)
    float* out = (float*)d_out;                  // (16384, 2048)

    char* ws = (char*)d_ws;
    const size_t bMatBytes = (size_t)OUT_F * IN_F * sizeof(__bf16);    // 4 MiB
    const size_t aMatBytes = (size_t)M_ROWS * IN_F * sizeof(__bf16);   // 32 MiB
    const size_t smallEnd  = 2 * bMatBytes + 2 * OUT_F * sizeof(float);
    const size_t needFull  = smallEnd + 2 * aMatBytes;

    __bf16* Bneg  = (__bf16*)(ws);
    __bf16* Bmean = (__bf16*)(ws + bMatBytes);
    float*  mcm   = (float*)(ws + 2 * bMatBytes);
    float*  tanhs = mcm + OUT_F;

    gauss_prep<<<OUT_F, 256, 0, stream>>>(diag, mean, scale, Bneg, Bmean, mcm, tanhs);

    dim3 grid(M_ROWS / 128, OUT_F / 64);   // 128 x 32 blocks, 8 waves each

    if (ws_size >= needFull) {
        __bf16* Azz = (__bf16*)(ws + smallEnd);
        __bf16* Az  = (__bf16*)(ws + smallEnd + aMatBytes);
        const int totalElems = M_ROWS * IN_F;
        gauss_prep_a<<<totalElems / (256 * 8), 256, 0, stream>>>(z, Azz, Az);
#if USE_TDM
        gauss_gemm_tdm<<<grid, 256, 0, stream>>>(Azz, Az, Bneg, Bmean, mcm, tanhs, out);
#else
        gauss_gemm_pre<<<grid, 256, 0, stream>>>(Azz, Az, Bneg, Bmean, mcm, tanhs, out);
#endif
    } else {
        gauss_gemm_conv<<<grid, 256, 0, stream>>>(z, Bneg, Bmean, mcm, tanhs, out);
    }
}
